// FusedMoEModularKernel_42434276884975
// MI455X (gfx1250) — matrix operand order repaired and checked by reading, not compiled, not run on hardware
//
#include <hip/hip_runtime.h>
#include <hip/hip_bf16.h>
#include <stdint.h>
#include <stddef.h>

// Problem constants (match reference)
#define E_      8
#define M_      4096
#define K_      2048
#define INTER_  5632
#define N_      (2 * INTER_)
#define TOPK_   2
#define SLOTS_  (M_ * TOPK_)   // 8192 routed (token, expert) pairs
#define CAP_    SLOTS_         // worst-case tokens routed to one expert

typedef __bf16 bf16_t;
typedef __attribute__((ext_vector_type(16))) __bf16 v16bf;
typedef __attribute__((ext_vector_type(8)))  __bf16 v8bf;
typedef __attribute__((ext_vector_type(8)))  float  v8f;

// ---- workspace layout (bytes) ----
static constexpr size_t CNT_OFF  = 0;                              // 8 x i32
static constexpr size_t LIST_OFF = 256;                            // 8*8192 x i32
static constexpr size_t ABF_OFF  = (size_t)1 << 20;                // a1 bf16: 16 MB
static constexpr size_t ACT_OFF  = ABF_OFF + ((size_t)20 << 20);   // act bf16: 92 MB
static constexpr size_t O_OFF    = ACT_OFF + ((size_t)96 << 20);   // slot out f32: 64 MB
// total ~ 181 MB

// LDS A-tile row stride (elements): +8 bf16 (16B) pad rotates bank groups per row
#define ASTRIDE_ (K_ + 8)

// ---------------- routing ----------------
__global__ void moe_route_init_k(int* __restrict__ cnt) {
    if (threadIdx.x < E_) cnt[threadIdx.x] = 0;
}

__global__ void moe_route_k(const long long* __restrict__ ids,
                            int* __restrict__ cnt, int* __restrict__ list) {
    int i = blockIdx.x * blockDim.x + threadIdx.x;   // slot id = m*TOPK + t
    if (i >= SLOTS_) return;
    int e = (int)ids[i];
    int pos = atomicAdd(&cnt[e], 1);
    list[e * CAP_ + pos] = i;
}

// ---------------- a1 f32 -> bf16 ----------------
__global__ void moe_cvt_a_k(const float* __restrict__ a, bf16_t* __restrict__ abf) {
    int idx = blockIdx.x * blockDim.x + threadIdx.x;   // over M*K/4
    float4 f = ((const float4*)a)[idx];
    union { bf16_t h[4]; uint2 u; } p;
    p.h[0] = (bf16_t)f.x; p.h[1] = (bf16_t)f.y;
    p.h[2] = (bf16_t)f.z; p.h[3] = (bf16_t)f.w;
    ((uint2*)abf)[idx] = p.u;
}

// ---------------- CDNA5 async global->LDS copy (16B per lane) ----------------
__device__ __forceinline__ void async_cp16(uint32_t lds_off, const void* gptr) {
    asm volatile("global_load_async_to_lds_b128 %0, %1, off"
                 :: "v"(lds_off), "v"((uint64_t)(uintptr_t)gptr)
                 : "memory");
}
__device__ __forceinline__ void wait_async0() {
    asm volatile("s_wait_asynccnt 0x0" ::: "memory");
}

// ---------------- WMMA fragment helpers ----------------
// A (16x32 bf16): lane<16 -> row=lane, K=0..15; lane>=16 -> row=lane-16, K=16..31
__device__ __forceinline__ v16bf load_a_frag(const bf16_t* __restrict__ row, int kb) {
    v8bf lo = *(const v8bf*)(row + 2 * kb);    // kb = 0 or 8 -> K start 0 or 16
    v8bf hi = *(const v8bf*)(row + 2 * kb + 8);
    return __builtin_shufflevector(lo, hi, 0, 1, 2, 3, 4, 5, 6, 7,
                                           8, 9, 10, 11, 12, 13, 14, 15);
}

// B (32x16 bf16): lane holds column (lane&15), K run = half*16..half*16+15 contiguous.
__device__ __forceinline__ v16bf load_b_frag_f32(const float* __restrict__ src) {
    v16bf b;
#pragma unroll
    for (int j = 0; j < 16; ++j) b[j] = (bf16_t)src[j];
    return b;
}

__device__ __forceinline__ float fast_silu(float g) {
    return g * __builtin_amdgcn_rcpf(1.0f + __expf(-g));
}

// ---------------- GEMM1 + SwiGLU ----------------
// Each 256-thread WG: 16 tokens x 256 inter-neurons (8 waves x 32 cols, gate+up).
__global__ void __launch_bounds__(256)
moe_gemm1_silu_k(const bf16_t* __restrict__ abf, const float* __restrict__ w1,
                 const int* __restrict__ cnt, const int* __restrict__ list,
                 bf16_t* __restrict__ act) {
    const int e  = blockIdx.z;
    const int mt = blockIdx.y;
    const int cnt_e = cnt[e];
    if (mt * 16 >= cnt_e) return;

    __shared__ __align__(16) bf16_t a_tile[16 * ASTRIDE_];   // ~64.3 KB
    __shared__ int rows_lds[16];

    if (threadIdx.x < 16) {
        int pos = mt * 16 + (int)threadIdx.x;
        rows_lds[threadIdx.x] = (pos < cnt_e) ? list[e * CAP_ + pos] : 0;
    }
    __syncthreads();

    // ---- async-stage the 16 x K A tile (bf16) into LDS ----
    {
        int r = threadIdx.x >> 4;             // row 0..15
        int c = threadIdx.x & 15;             // 16B chunk lane 0..15
        int tok = rows_lds[r] >> 1;           // slot -> token (m = slot / TOPK)
        const bf16_t* gsrc = abf + (size_t)tok * K_;
        uint32_t lbase = (uint32_t)(uintptr_t)(a_tile) + (uint32_t)(r * ASTRIDE_ * 2);
#pragma unroll
        for (int i = 0; i < K_ / 128; ++i) {  // 16 x 16B per thread
            int eoff = c * 8 + i * 128;       // element offset within row
            async_cp16(lbase + (uint32_t)(eoff * 2), gsrc + eoff);
        }
        wait_async0();
    }
    __syncthreads();

    const int wave = threadIdx.x >> 5;
    const int lane = threadIdx.x & 31;
    const int half = lane >> 4;
    const int lmod = lane & 15;
    const int kb   = half * 8;
    const int n0   = (blockIdx.x * 8 + wave) * 32;   // 32 inter-neurons per wave

    const bf16_t* arow = a_tile + (size_t)lmod * ASTRIDE_;
    const float*  bg0  = w1 + ((size_t)e * N_ + (size_t)(n0 + lmod)) * K_ + half * 16;
    const float*  bg1  = bg0 + (size_t)16 * K_;
    const float*  bu0  = w1 + ((size_t)e * N_ + (size_t)(INTER_ + n0 + lmod)) * K_ + half * 16;
    const float*  bu1  = bu0 + (size_t)16 * K_;

    v8f ag0 = {}, au0 = {}, ag1 = {}, au1 = {};
    for (int k0 = 0; k0 < K_; k0 += 32) {
        v16bf a  = load_a_frag(arow + k0, kb);       // ds_load_b128 x2
        v16bf g0 = load_b_frag_f32(bg0 + k0);
        v16bf u0 = load_b_frag_f32(bu0 + k0);
        v16bf g1 = load_b_frag_f32(bg1 + k0);
        v16bf u1 = load_b_frag_f32(bu1 + k0);
        ag0 = __builtin_amdgcn_wmma_f32_16x16x32_bf16(false, a, false, g0, (short)0, ag0, false, false);
        au0 = __builtin_amdgcn_wmma_f32_16x16x32_bf16(false, a, false, u0, (short)0, au0, false, false);
        ag1 = __builtin_amdgcn_wmma_f32_16x16x32_bf16(false, a, false, g1, (short)0, ag1, false, false);
        au1 = __builtin_amdgcn_wmma_f32_16x16x32_bf16(false, a, false, u1, (short)0, au1, false, false);
    }

#pragma unroll
    for (int v = 0; v < 8; ++v) {
        int m = v + half * 8;                        // D layout: M = vgpr + 8*half
        if (mt * 16 + m < cnt_e) {
            int slot = rows_lds[m];
            bf16_t* dst = act + (size_t)slot * INTER_ + (size_t)(n0 + lmod);
            dst[0]  = (bf16_t)(fast_silu(ag0[v]) * au0[v]);
            dst[16] = (bf16_t)(fast_silu(ag1[v]) * au1[v]);
        }
    }
}

// ---------------- GEMM2 ----------------
// Each 256-thread WG: 16 slots x 256 hidden cols (8 waves x 32 cols).
__global__ void __launch_bounds__(256)
moe_gemm2_k(const bf16_t* __restrict__ act, const float* __restrict__ w2,
            const int* __restrict__ cnt, const int* __restrict__ list,
            float* __restrict__ o) {
    const int e  = blockIdx.z;
    const int mt = blockIdx.y;
    const int cnt_e = cnt[e];
    if (mt * 16 >= cnt_e) return;

    __shared__ int rows_lds[16];
    if (threadIdx.x < 16) {
        int pos = mt * 16 + (int)threadIdx.x;
        rows_lds[threadIdx.x] = (pos < cnt_e) ? list[e * CAP_ + pos] : 0;
    }
    __syncthreads();

    const int wave = threadIdx.x >> 5;
    const int lane = threadIdx.x & 31;
    const int half = lane >> 4;
    const int lmod = lane & 15;
    const int kb   = half * 8;
    const int c0   = (blockIdx.x * 8 + wave) * 32;   // 32 hidden cols per wave

    const int slot_m = rows_lds[lmod];
    const bf16_t* arow = act + (size_t)slot_m * INTER_;
    const float*  bw0  = w2 + ((size_t)e * K_ + (size_t)(c0 + lmod)) * INTER_ + half * 16;
    const float*  bw1  = bw0 + (size_t)16 * INTER_;

    v8f acc0 = {}, acc1 = {};
    for (int n0 = 0; n0 < INTER_; n0 += 32) {
        v16bf a  = load_a_frag(arow + n0, kb);
        v16bf b0 = load_b_frag_f32(bw0 + n0);
        v16bf b1 = load_b_frag_f32(bw1 + n0);
        acc0 = __builtin_amdgcn_wmma_f32_16x16x32_bf16(false, a, false, b0, (short)0, acc0, false, false);
        acc1 = __builtin_amdgcn_wmma_f32_16x16x32_bf16(false, a, false, b1, (short)0, acc1, false, false);
    }

#pragma unroll
    for (int v = 0; v < 8; ++v) {
        int m = v + half * 8;
        if (mt * 16 + m < cnt_e) {
            int slot = rows_lds[m];
            float* dst = o + (size_t)slot * K_ + (size_t)(c0 + lmod);
            dst[0]  = acc0[v];
            dst[16] = acc1[v];
        }
    }
}

// ---------------- combine (deterministic, no atomics) ----------------
__global__ void moe_combine_k(const float* __restrict__ tw, const float* __restrict__ o,
                              float* __restrict__ out) {
    int idx = blockIdx.x * blockDim.x + threadIdx.x;   // over M*K/4
    int m  = idx / (K_ / 4);
    int kq = idx - m * (K_ / 4);
    float w0 = tw[m * 2 + 0];
    float w1 = tw[m * 2 + 1];
    const float4* o4 = (const float4*)o;
    float4 x = o4[(size_t)(m * 2 + 0) * (K_ / 4) + kq];
    float4 y = o4[(size_t)(m * 2 + 1) * (K_ / 4) + kq];
    float4 r;
    r.x = w0 * x.x + w1 * y.x;
    r.y = w0 * x.y + w1 * y.y;
    r.z = w0 * x.z + w1 * y.z;
    r.w = w0 * x.w + w1 * y.w;
    ((float4*)out)[idx] = r;
}

extern "C" void kernel_launch(void* const* d_in, const int* in_sizes, int n_in,
                              void* d_out, int out_size, void* d_ws, size_t ws_size,
                              hipStream_t stream) {
    (void)in_sizes; (void)n_in; (void)out_size; (void)ws_size;
    const float*     a1  = (const float*)d_in[0];
    const float*     w1  = (const float*)d_in[1];
    const float*     w2  = (const float*)d_in[2];
    const float*     tw  = (const float*)d_in[3];
    const long long* ids = (const long long*)d_in[4];   // int64 topk_ids

    char* ws = (char*)d_ws;
    int*    cnt  = (int*)(ws + CNT_OFF);
    int*    list = (int*)(ws + LIST_OFF);
    bf16_t* abf  = (bf16_t*)(ws + ABF_OFF);
    bf16_t* act  = (bf16_t*)(ws + ACT_OFF);
    float*  o    = (float*)(ws + O_OFF);
    float*  out  = (float*)d_out;

    hipLaunchKernelGGL(moe_route_init_k, dim3(1), dim3(64), 0, stream, cnt);
    hipLaunchKernelGGL(moe_route_k, dim3((SLOTS_ + 255) / 256), dim3(256), 0, stream,
                       ids, cnt, list);
    hipLaunchKernelGGL(moe_cvt_a_k, dim3((M_ * K_ / 4) / 256), dim3(256), 0, stream,
                       a1, abf);
    // GEMM1: x = INTER/(8 waves * 32 cols) = 22, y = worst-case m-tiles, z = expert
    hipLaunchKernelGGL(moe_gemm1_silu_k, dim3(INTER_ / 256, CAP_ / 16, E_), dim3(256),
                       0, stream, abf, w1, cnt, list, act);
    // GEMM2: x = K/256 = 8
    hipLaunchKernelGGL(moe_gemm2_k, dim3(K_ / 256, CAP_ / 16, E_), dim3(256),
                       0, stream, act, w2, cnt, list, o);
    hipLaunchKernelGGL(moe_combine_k, dim3((M_ * K_ / 4) / 256), dim3(256), 0, stream,
                       tw, o, out);
}
